// P4TransformerAnchor_76261439308370
// MI455X (gfx1250) — matrix operand order repaired and checked by reading, not compiled, not run on hardware
//
#include <hip/hip_runtime.h>
#include <hip/hip_bf16.h>
#include <stdint.h>

// ---------------------------------------------------------------------------
// P4Transformer-with-anchors forward, CDNA5 (gfx1250, wave32, WMMA) version.
//
// Sizes (from reference): B=8, L=5, N=4096, N_ANCH=64 -> NTOT=4160, M=260,
// nF=3 frames -> SEQ=780 tokens, DIM=512, DEPTH=5, HEADS=8, DHEAD=64,
// MLP=1024, OUT=1088, K(ball)=32, RADIUS^2=0.49.
//
// All GEMMs run on v_wmma_f32_16x16x32_bf16 (bf16 in / f32 accumulate).
// Weights are transposed+converted to bf16 once per launch so every
// weight-stationary GEMM uses contiguous (row-major-in-K) loads.
// Attention is 3-pass; the 156MB score tensor fits in the 192MB L2.
// Workspace use: ~315 MB.
// ---------------------------------------------------------------------------

#define B_    8
#define L_    5
#define N_    4096
#define NANCH 64
#define NTOT  4160
#define MM    260
#define NF    3
#define SEQ   780
#define DIM_  512
#define DEPTH_ 5
#define HEADS_ 8
#define DHEAD 64
#define MLP_  1024
#define OUTD  1088
#define KNN   32
#define RAD2  0.49f

typedef __attribute__((ext_vector_type(16))) __bf16 bf16x16;
typedef __attribute__((ext_vector_type(8)))  float  floatx8;

__device__ __forceinline__ unsigned short f2bf(float f) {
  uint32_t u = __float_as_uint(f);
  u += 0x7FFFu + ((u >> 16) & 1u);           // round-to-nearest-even
  return (unsigned short)(u >> 16);
}

// ---------------------------------------------------------------------------
// 1) Build padded point clouds: pts[b,l,i] = points[b,l,i] (i<N) else anchor.
// ---------------------------------------------------------------------------
__global__ void k_build_pts(const float* __restrict__ points, float* __restrict__ pts) {
  int i = blockIdx.x * blockDim.x + threadIdx.x;
  const int tot = B_ * L_ * NTOT;
  if (i >= tot) return;
  int p = i % NTOT, bl = i / NTOT;
  float x, y, z;
  if (p < N_) {
    const float* s = points + ((long long)bl * N_ + p) * 3;
    x = s[0]; y = s[1]; z = s[2];
  } else {
    int a = p - N_;
    x = -1.5f + (float)(a >> 4) * 1.0f;
    y = -1.5f + (float)((a >> 2) & 3) * 1.0f;
    z = -1.5f + (float)(a & 3) * 1.0f;
  }
  float* d = pts + (long long)i * 3;
  d[0] = x; d[1] = y; d[2] = z;
}

// ---------------------------------------------------------------------------
// 2) Farthest point sampling: one block per (b, frame f). Frame = pts[:,2f].
//    Matches jax scan: emit `last` first, then min-update + argmax.
// ---------------------------------------------------------------------------
__global__ void __launch_bounds__(256) k_fps(const float* __restrict__ pts,
                                             int* __restrict__ a_idx,
                                             float* __restrict__ a_xyz) {
  __shared__ float pp[NTOT * 3];
  __shared__ float sv[256];
  __shared__ int   si[256];
  __shared__ int   slast;
  const int tid = threadIdx.x;
  const int bf = blockIdx.x;
  const int b = bf / NF, f = bf % NF;
  const float* P = pts + ((long long)(b * L_ + 2 * f)) * NTOT * 3;
  for (int i = tid; i < NTOT * 3; i += 256) pp[i] = P[i];
  __syncthreads();

  const int CH = 17;                 // 256*17 >= 4160
  float dist[CH];
#pragma unroll
  for (int j = 0; j < CH; ++j) dist[j] = 1e10f;
  const int base = tid * CH;
  int last = 0;
  int*   oI = a_idx + (long long)bf * MM;
  float* oX = a_xyz + (long long)bf * MM * 3;

  for (int m = 0; m < MM; ++m) {
    if (tid == 0) {
      oI[m] = last;
      oX[m * 3 + 0] = pp[last * 3 + 0];
      oX[m * 3 + 1] = pp[last * 3 + 1];
      oX[m * 3 + 2] = pp[last * 3 + 2];
    }
    const float lx = pp[last * 3], ly = pp[last * 3 + 1], lz = pp[last * 3 + 2];
    float bv = -1.0f; int bi = 0x7fffffff;
    for (int j = 0; j < CH; ++j) {
      int g = base + j;
      if (g < NTOT) {
        float dx = pp[g * 3] - lx, dy = pp[g * 3 + 1] - ly, dz = pp[g * 3 + 2] - lz;
        float d = dx * dx + dy * dy + dz * dz;
        if (d < dist[j]) dist[j] = d;
        if (dist[j] > bv) { bv = dist[j]; bi = g; }   // strict > keeps lowest idx
      }
    }
    sv[tid] = bv; si[tid] = bi;
    __syncthreads();
    for (int s = 128; s > 0; s >>= 1) {
      if (tid < s) {
        if (sv[tid + s] > sv[tid] ||
            (sv[tid + s] == sv[tid] && si[tid + s] < si[tid])) {
          sv[tid] = sv[tid + s]; si[tid] = si[tid + s];
        }
      }
      __syncthreads();
    }
    if (tid == 0) slast = si[0];
    __syncthreads();
    last = slast;
  }
}

// ---------------------------------------------------------------------------
// 3) Ball query + point conv + max over K + max over time window + pos embed.
//    One block per (b, f, m). Writes x[b, f*M+m, :] fp32.
//    feat[d] = max_k (p_k . (cf_d+cd_d[:3])) - a.(cd_d[:3]) + (i-t)*cd_d[3]
// ---------------------------------------------------------------------------
__global__ void __launch_bounds__(256) k_group(
    const float* __restrict__ pts, const float* __restrict__ a_xyz,
    const float* __restrict__ conv_d_w, const float* __restrict__ conv_f_w,
    const float* __restrict__ pos_w, const float* __restrict__ pos_b,
    float* __restrict__ x) {
  __shared__ float gp[KNN * 3];
  __shared__ int sc[2][256];
  const int tid = threadIdx.x;
  const int blk = blockIdx.x;
  const int m = blk % MM;
  const int bf = blk / MM;
  const int f = bf % NF, b = bf / NF;
  const float ax = a_xyz[((long long)bf * MM + m) * 3 + 0];
  const float ay = a_xyz[((long long)bf * MM + m) * 3 + 1];
  const float az = a_xyz[((long long)bf * MM + m) * 3 + 2];

  float w0[2], w1[2], w2[2], cdx[2], cdy[2], cdz[2], cdt[2], best[2];
#pragma unroll
  for (int c = 0; c < 2; ++c) {
    int d = tid + c * 256;
    w0[c] = conv_f_w[d * 3 + 0] + conv_d_w[d * 4 + 0];
    w1[c] = conv_f_w[d * 3 + 1] + conv_d_w[d * 4 + 1];
    w2[c] = conv_f_w[d * 3 + 2] + conv_d_w[d * 4 + 2];
    cdx[c] = conv_d_w[d * 4 + 0]; cdy[c] = conv_d_w[d * 4 + 1];
    cdz[c] = conv_d_w[d * 4 + 2]; cdt[c] = conv_d_w[d * 4 + 3];
    best[c] = -3.4e38f;
  }

  for (int io = -1; io <= 1; ++io) {
    int l = 2 * f + io; l = l < 0 ? 0 : (l > L_ - 1 ? L_ - 1 : l);
    const float* P = pts + ((long long)(b * L_ + l)) * NTOT * 3;
    // ordered ball query: first 32 indices (ascending) with d2 < r^2
    const int CH = 17;
    int lidx[CH]; int cnt = 0;
    for (int j = 0; j < CH; ++j) {
      int g = tid * CH + j;
      if (g < NTOT) {
        float dx = P[g * 3] - ax, dy = P[g * 3 + 1] - ay, dz = P[g * 3 + 2] - az;
        if (dx * dx + dy * dy + dz * dz < RAD2) lidx[cnt++] = g;
      }
    }
    // Hillis-Steele inclusive scan of counts (8 steps)
    sc[0][tid] = cnt;
    __syncthreads();
    int src = 0;
#pragma unroll
    for (int d = 1; d < 256; d <<= 1) {
      int t = sc[src][tid] + ((tid >= d) ? sc[src][tid - d] : 0);
      sc[src ^ 1][tid] = t;
      src ^= 1;
      __syncthreads();
    }
    int start = sc[src][tid] - cnt;
    int total = sc[src][255];
    for (int q = 0; q < cnt; ++q) {
      int slot = start + q;
      if (slot < KNN) {
        int g = lidx[q];
        gp[slot * 3 + 0] = P[g * 3 + 0];
        gp[slot * 3 + 1] = P[g * 3 + 1];
        gp[slot * 3 + 2] = P[g * 3 + 2];
      }
    }
    __syncthreads();
    if (tid == 0 && total < KNN) {       // fill with first hit, else point 0
      float fx, fy, fz;
      if (total > 0) { fx = gp[0]; fy = gp[1]; fz = gp[2]; }
      else { fx = P[0]; fy = P[1]; fz = P[2]; }
      for (int s = total; s < KNN; ++s) { gp[s*3] = fx; gp[s*3+1] = fy; gp[s*3+2] = fz; }
    }
    __syncthreads();
    const float dt = (float)io;
#pragma unroll
    for (int c = 0; c < 2; ++c) {
      float mk = -3.4e38f;
      for (int k = 0; k < KNN; ++k) {
        float v = gp[k*3] * w0[c] + gp[k*3+1] * w1[c] + gp[k*3+2] * w2[c];
        mk = fmaxf(mk, v);
      }
      float cst = -(ax * cdx[c] + ay * cdy[c] + az * cdz[c]) + dt * cdt[c];
      best[c] = fmaxf(best[c], mk + cst);
    }
    __syncthreads();                     // before gp/sc reuse
  }

  const float tval = (float)(f + 1);
#pragma unroll
  for (int c = 0; c < 2; ++c) {
    int d = tid + c * 256;
    float pe = ax * pos_w[d*4+0] + ay * pos_w[d*4+1] + az * pos_w[d*4+2]
             + tval * pos_w[d*4+3] + pos_b[d];
    x[((long long)b * SEQ + f * MM + m) * DIM_ + d] = best[c] + pe;
  }
}

// ---------------------------------------------------------------------------
// 4) LayerNorm over DIM_=512, fp32 in -> bf16 out. One block per row.
// ---------------------------------------------------------------------------
__global__ void __launch_bounds__(256) k_ln(const float* __restrict__ X,
                                            const float* __restrict__ g,
                                            const float* __restrict__ bta,
                                            unsigned short* __restrict__ Y) {
  __shared__ float red[256];
  const int r = blockIdx.x, tid = threadIdx.x;
  const float* xr = X + (long long)r * DIM_;
  float v0 = xr[tid], v1 = xr[tid + 256];
  red[tid] = v0 + v1;
  __syncthreads();
  for (int s = 128; s > 0; s >>= 1) { if (tid < s) red[tid] += red[tid + s]; __syncthreads(); }
  float mu = red[0] * (1.0f / DIM_);
  __syncthreads();
  float a0 = v0 - mu, a1 = v1 - mu;
  red[tid] = a0 * a0 + a1 * a1;
  __syncthreads();
  for (int s = 128; s > 0; s >>= 1) { if (tid < s) red[tid] += red[tid + s]; __syncthreads(); }
  float rs = rsqrtf(red[0] * (1.0f / DIM_) + 1e-5f);
  unsigned short* yr = Y + (long long)r * DIM_;
  yr[tid]       = f2bf(a0 * rs * g[tid]       + bta[tid]);
  yr[tid + 256] = f2bf(a1 * rs * g[tid + 256] + bta[tid + 256]);
}

// ---------------------------------------------------------------------------
// 5) Row softmax over SEQ=780: fp32 scores -> bf16 probs. One block per row.
// ---------------------------------------------------------------------------
__global__ void __launch_bounds__(256) k_softmax(const float* __restrict__ S,
                                                 unsigned short* __restrict__ P) {
  __shared__ float red[256];
  const long long r = blockIdx.x;
  const float* sr = S + r * SEQ;
  unsigned short* pr = P + r * SEQ;
  const int tid = threadIdx.x;
  float v[4]; float mx = -3.4e38f;
#pragma unroll
  for (int j = 0; j < 4; ++j) {
    int c = tid + j * 256;
    v[j] = (c < SEQ) ? sr[c] : -3.4e38f;
    mx = fmaxf(mx, v[j]);
  }
  red[tid] = mx; __syncthreads();
  for (int s = 128; s > 0; s >>= 1) { if (tid < s) red[tid] = fmaxf(red[tid], red[tid + s]); __syncthreads(); }
  mx = red[0]; __syncthreads();
  float sum = 0.0f;
#pragma unroll
  for (int j = 0; j < 4; ++j) { v[j] = expf(v[j] - mx); sum += v[j]; }
  red[tid] = sum; __syncthreads();
  for (int s = 128; s > 0; s >>= 1) { if (tid < s) red[tid] += red[tid + s]; __syncthreads(); }
  float inv = 1.0f / red[0];
#pragma unroll
  for (int j = 0; j < 4; ++j) {
    int c = tid + j * 256;
    if (c < SEQ) pr[c] = f2bf(v[j] * inv);
  }
}

// ---------------------------------------------------------------------------
// 6) Generic batched bf16 WMMA GEMM.
//    C[z][M,N] = act(alpha * A[z] @ W[z] + bias) (+ R[z]); dual f32/bf16 out.
//    Batch offsets: off = (z/zdiv)*s1 + (z%zdiv)*s2 (element units).
//    W access: wtrans ? W[n*ldw + k] : W[k*ldw + n].
//    Block: 128x128 tile, 8 waves of 64x32; K stepped by 32.
//
//    Loader strategy (branchless): rows/cols beyond M/N are *clamped* to a
//    valid address -- an out-of-range A row / B col only pollutes C rows/cols
//    that are never stored, so no masking is needed. Only the K-dim tail is
//    zero-filled (selects, final K-step only). Interior steps use unguarded
//    8-byte vector loads (all lda/ldw are multiples of 4 elements).
//
//    LDS: A-tile and transposed-B-tile, both [128 rows][32 bf16], 20-dword
//    row stride; fragment loads are two ds_load_b128 per frag matching the
//    documented wave32 WMMA A/B VGPR layouts.
// ---------------------------------------------------------------------------
__device__ __forceinline__ bf16x16 frag_from(const uint32_t* rowp, int off0, int off1) {
  union { uint4 q[2]; bf16x16 v; } u;
  u.q[0] = *(const uint4*)(rowp + off0);
  u.q[1] = *(const uint4*)(rowp + off1);
  return u.v;
}

__global__ void __launch_bounds__(256) k_gemm(
    const unsigned short* __restrict__ A, long long sA1, long long sA2, int lda,
    const unsigned short* __restrict__ W, long long sW1, long long sW2, int ldw, int wtrans,
    const float* __restrict__ bias,
    const float* R, long long sR1, long long sR2,
    float* Cf, long long sCf1, long long sCf2,
    unsigned short* Cb, long long sCb1, long long sCb2,
    int ldc, int Mdim, int Ndim, int Kdim, int zdiv, float alpha, int act) {
  __shared__ uint32_t lds[5120];        // A: [0,2560), B^T: [2560,5120)
  const int z = blockIdx.z, zq = z / zdiv, zr = z % zdiv;
  const unsigned short* Ab = A + zq * sA1 + zr * sA2;
  const unsigned short* Wb = W + zq * sW1 + zr * sW2;
  const float* Rb = R ? (R + zq * sR1 + zr * sR2) : nullptr;
  float* Cfb = Cf ? (Cf + zq * sCf1 + zr * sCf2) : nullptr;
  unsigned short* Cbb = Cb ? (Cb + zq * sCb1 + zr * sCb2) : nullptr;

  const int m0 = blockIdx.y * 128, n0 = blockIdx.x * 128;
  const int tid = threadIdx.x;
  const int lane = tid & 31, wv = tid >> 5;
  const int wm = (wv & 1) * 64, wn = (wv >> 1) * 32;
  const int khi = lane >> 4, r = lane & 15;

  floatx8 acc[4][2];
  const floatx8 zero8 = {0.f, 0.f, 0.f, 0.f, 0.f, 0.f, 0.f, 0.f};
#pragma unroll
  for (int fm = 0; fm < 4; ++fm)
#pragma unroll
    for (int fn = 0; fn < 2; ++fn) acc[fm][fn] = zero8;

  const int row = tid >> 1, hf = tid & 1;
  const int arow_i = (m0 + row) < Mdim ? (m0 + row) : (Mdim - 1);   // clamp
  const int nc     = (n0 + row) < Ndim ? (n0 + row) : (Ndim - 1);   // clamp
  const unsigned short* arow = Ab + (long long)arow_i * lda;
  const unsigned short* wrow = Wb + (long long)nc * ldw;            // wtrans rows

  const int nK = (Kdim + 31) >> 5;
  for (int kt = 0; kt < nK; ++kt) {
    const int k0 = kt << 5;
    uint2 av[4], bv[4];
    if (k0 + 32 <= Kdim) {
      // -------- fast path: unguarded 8-byte vector loads --------
      const unsigned short* pa = arow + k0 + hf * 16;
#pragma unroll
      for (int j = 0; j < 4; ++j) av[j] = *(const uint2*)(pa + j * 4);
      if (wtrans) {
        const unsigned short* pb = wrow + k0 + hf * 16;
#pragma unroll
        for (int j = 0; j < 4; ++j) bv[j] = *(const uint2*)(pb + j * 4);
      } else {
#pragma unroll
        for (int j = 0; j < 4; ++j) {
          int kk = k0 + hf * 16 + j * 4;
          uint32_t e0 = Wb[(long long)(kk + 0) * ldw + nc];
          uint32_t e1 = Wb[(long long)(kk + 1) * ldw + nc];
          uint32_t e2 = Wb[(long long)(kk + 2) * ldw + nc];
          uint32_t e3 = Wb[(long long)(kk + 3) * ldw + nc];
          bv[j].x = e0 | (e1 << 16);
          bv[j].y = e2 | (e3 << 16);
        }
      }
    } else {
      // -------- K-tail: clamped loads + zero-select (branchless) --------
#pragma unroll
      for (int j = 0; j < 4; ++j) {
        int kk = k0 + hf * 16 + j * 4;
        int c0 = (kk     < Kdim) ? kk     : (Kdim - 2);
        int c1 = (kk + 2 < Kdim) ? kk + 2 : (Kdim - 2);
        uint32_t lo = *(const uint32_t*)(arow + c0);
        uint32_t hi = *(const uint32_t*)(arow + c1);
        av[j].x = (kk     < Kdim) ? lo : 0u;
        av[j].y = (kk + 2 < Kdim) ? hi : 0u;
        if (wtrans) {
          uint32_t wl = *(const uint32_t*)(wrow + c0);
          uint32_t wh = *(const uint32_t*)(wrow + c1);
          bv[j].x = (kk     < Kdim) ? wl : 0u;
          bv[j].y = (kk + 2 < Kdim) ? wh : 0u;
        } else {
          int q0 = (kk     < Kdim) ? kk     : (Kdim - 1);
          int q1 = (kk + 1 < Kdim) ? kk + 1 : (Kdim - 1);
          int q2 = (kk + 2 < Kdim) ? kk + 2 : (Kdim - 1);
          int q3 = (kk + 3 < Kdim) ? kk + 3 : (Kdim - 1);
          uint32_t e0 = Wb[(long long)q0 * ldw + nc];
          uint32_t e1 = Wb[(long long)q1 * ldw + nc];
          uint32_t e2 = Wb[(long long)q2 * ldw + nc];
          uint32_t e3 = Wb[(long long)q3 * ldw + nc];
          uint32_t xw = e0 | (e1 << 16), yw = e2 | (e3 << 16);
          xw = (kk + 1 < Kdim) ? xw : (xw & 0x0000FFFFu);
          xw = (kk     < Kdim) ? xw : 0u;
          yw = (kk + 3 < Kdim) ? yw : (yw & 0x0000FFFFu);
          yw = (kk + 2 < Kdim) ? yw : 0u;
          bv[j].x = xw; bv[j].y = yw;
        }
      }
    }
    // stage to LDS (b64 stores, batched)
    {
      uint32_t* la = &lds[row * 20 + hf * 8];
      uint32_t* lb = &lds[2560 + row * 20 + hf * 8];
#pragma unroll
      for (int j = 0; j < 4; ++j) *(uint2*)(la + j * 2) = av[j];
#pragma unroll
      for (int j = 0; j < 4; ++j) *(uint2*)(lb + j * 2) = bv[j];
    }
    if (kt + 1 < nK)
      __builtin_prefetch(arow + k0 + 32 + hf * 16, 0, 1);   // global_prefetch_b8
    __syncthreads();

    bf16x16 af[4];
#pragma unroll
    for (int fm = 0; fm < 4; ++fm)
      af[fm] = frag_from(&lds[(wm + fm * 16 + r) * 20], khi * 4, 8 + khi * 4);
#pragma unroll
    for (int fn = 0; fn < 2; ++fn) {
      bf16x16 bfr = frag_from(&lds[2560 + (wn + fn * 16 + r) * 20], khi * 4, 8 + khi * 4);
#pragma unroll
      for (int fm = 0; fm < 4; ++fm)
        acc[fm][fn] = __builtin_amdgcn_wmma_f32_16x16x32_bf16(
            false, af[fm], false, bfr, (short)0, acc[fm][fn], false, false);
    }
    __syncthreads();
  }

  // Epilogue. C layout: lane 0-15 -> N=lane, M=e; lane 16-31 -> N=lane-16, M=e+8.
#pragma unroll
  for (int fm = 0; fm < 4; ++fm) {
#pragma unroll
    for (int fn = 0; fn < 2; ++fn) {
      const int n = n0 + wn + fn * 16 + r;
      if (n >= Ndim) continue;
      const float bval = bias ? bias[n] : 0.0f;
#pragma unroll
      for (int e = 0; e < 8; ++e) {
        const int mr = m0 + wm + fm * 16 + e + 8 * khi;
        if (mr >= Mdim) continue;
        float v = acc[fm][fn][e] * alpha + bval;
        if (act == 1) v = 0.5f * v * (1.0f + erff(v * 0.70710678118654752f));
        const long long ci = (long long)mr * ldc + n;
        if (Rb)  v += Rb[ci];
        if (Cfb) Cfb[ci] = v;
        if (Cbb) Cbb[ci] = f2bf(v);
      }
    }
  }
}

// ---------------------------------------------------------------------------
// 7) Max over sequence dim: x (B,SEQ,512) -> xh (B,512).
// ---------------------------------------------------------------------------
__global__ void k_seqmax(const float* __restrict__ X, float* __restrict__ Y) {
  int i = blockIdx.x * blockDim.x + threadIdx.x;
  if (i >= B_ * DIM_) return;
  int b = i / DIM_, d = i % DIM_;
  const float* p = X + (long long)b * SEQ * DIM_ + d;
  float m = -3.4e38f;
  for (int s = 0; s < SEQ; ++s) m = fmaxf(m, p[(long long)s * DIM_]);
  Y[i] = m;
}

// ---------------------------------------------------------------------------
// 8) Head output: per (b, group) softmax over 64 then @ anchor grid (64x3).
// ---------------------------------------------------------------------------
__global__ void __launch_bounds__(64) k_head_out(const float* __restrict__ H,
                                                 float* __restrict__ out) {
  __shared__ float pb[64];
  __shared__ float red[64];
  const int bg = blockIdx.x, tid = threadIdx.x;
  const float v = H[(long long)(bg / 17) * OUTD + (bg % 17) * 64 + tid];
  red[tid] = v; __syncthreads();
  for (int s = 32; s > 0; s >>= 1) { if (tid < s) red[tid] = fmaxf(red[tid], red[tid + s]); __syncthreads(); }
  float mx = red[0]; __syncthreads();
  float e = expf(v - mx);
  red[tid] = e; __syncthreads();
  for (int s = 32; s > 0; s >>= 1) { if (tid < s) red[tid] += red[tid + s]; __syncthreads(); }
  float sum = red[0]; __syncthreads();
  pb[tid] = e / sum; __syncthreads();
  if (tid < 3) {
    float acc = 0.0f;
    for (int i = 0; i < 64; ++i) {
      int q = (tid == 0) ? (i >> 4) : (tid == 1 ? ((i >> 2) & 3) : (i & 3));
      acc += pb[i] * (-1.5f + (float)q);
    }
    out[(long long)bg * 3 + tid] = acc;
  }
}

// ---------------------------------------------------------------------------
// 9) Transpose + convert weights: src (K x N) f32 -> dst (N x K) bf16.
//    Tiled 32x32 via LDS (coalesced on both sides). blockIdx.z = layer slice.
// ---------------------------------------------------------------------------
__global__ void __launch_bounds__(256) k_cvt_t(const float* __restrict__ src,
                                               unsigned short* __restrict__ dst,
                                               int K, int N) {
  __shared__ float t[32][33];
  const long long so = (long long)blockIdx.z * K * N;
  const int kb = blockIdx.y * 32, nb = blockIdx.x * 32;
  const int tx = threadIdx.x & 31, ty = threadIdx.x >> 5;   // 32 x 8
#pragma unroll
  for (int i = 0; i < 4; ++i) {
    int k = kb + ty + i * 8, n = nb + tx;
    t[ty + i * 8][tx] = (k < K && n < N) ? src[so + (long long)k * N + n] : 0.0f;
  }
  __syncthreads();
#pragma unroll
  for (int i = 0; i < 4; ++i) {
    int n = nb + ty + i * 8, k = kb + tx;
    if (n < N && k < K) dst[so + (long long)n * K + k] = f2bf(t[tx][ty + i * 8]);
  }
}

// ---------------------------------------------------------------------------
// Host orchestration
// ---------------------------------------------------------------------------
extern "C" void kernel_launch(void* const* d_in, const int* in_sizes, int n_in,
                              void* d_out, int out_size, void* d_ws, size_t ws_size,
                              hipStream_t stream) {
  const float* points     = (const float*)d_in[0];
  const float* conv_d_w   = (const float*)d_in[1];
  const float* conv_f_w   = (const float*)d_in[2];
  const float* pos_w      = (const float*)d_in[3];
  const float* pos_b      = (const float*)d_in[4];
  const float* ln1_g      = (const float*)d_in[5];
  const float* ln1_b      = (const float*)d_in[6];
  const float* qkv_w      = (const float*)d_in[7];
  const float* attn_out_w = (const float*)d_in[8];
  const float* attn_out_b = (const float*)d_in[9];
  const float* ln2_g      = (const float*)d_in[10];
  const float* ln2_b      = (const float*)d_in[11];
  const float* ff1_w      = (const float*)d_in[12];
  const float* ff1_b      = (const float*)d_in[13];
  const float* ff2_w      = (const float*)d_in[14];
  const float* ff2_b      = (const float*)d_in[15];
  const float* head_ln_g  = (const float*)d_in[16];
  const float* head_ln_b  = (const float*)d_in[17];
  const float* head1_w    = (const float*)d_in[18];
  const float* head1_b    = (const float*)d_in[19];
  const float* head2_w    = (const float*)d_in[20];
  const float* head2_b    = (const float*)d_in[21];

  char* ws = (char*)d_ws;
  size_t off = 0;
  auto carve = [&](size_t bytes) -> void* {
    void* p = ws + off;
    off += (bytes + 255) & ~(size_t)255;
    return p;
  };
  const long long ROWS = (long long)B_ * SEQ;          // 6240
  const long long SS   = (long long)SEQ * SEQ;         // 608400
  const int BH = B_ * HEADS_;                          // 64

  float*          pts    = (float*)         carve((size_t)B_ * L_ * NTOT * 3 * 4);
  int*            a_idx  = (int*)           carve((size_t)B_ * NF * MM * 4);
  float*          a_xyz  = (float*)         carve((size_t)B_ * NF * MM * 3 * 4);
  float*          xbuf   = (float*)         carve((size_t)ROWS * DIM_ * 4);
  unsigned short* xn     = (unsigned short*)carve((size_t)ROWS * DIM_ * 2);
  unsigned short* qkvb   = (unsigned short*)carve((size_t)ROWS * 1536 * 2);
  float*          scores = (float*)         carve((size_t)BH * SS * 4);      // 156 MB (fits L2)
  unsigned short* probs  = (unsigned short*)carve((size_t)BH * SS * 2);
  unsigned short* obuf   = (unsigned short*)carve((size_t)ROWS * DIM_ * 2);
  unsigned short* ffh    = (unsigned short*)carve((size_t)ROWS * MLP_ * 2);
  unsigned short* wqT    = (unsigned short*)carve((size_t)DEPTH_ * 512 * 1536 * 2);  // (1536,512) per layer
  unsigned short* woT    = (unsigned short*)carve((size_t)DEPTH_ * 512 * 512 * 2);   // (512,512)
  unsigned short* w1T    = (unsigned short*)carve((size_t)DEPTH_ * 512 * 1024 * 2);  // (1024,512)
  unsigned short* w2T    = (unsigned short*)carve((size_t)DEPTH_ * 1024 * 512 * 2);  // (512,1024)
  unsigned short* h1wT   = (unsigned short*)carve((size_t)512 * 1024 * 2);           // (1024,512)
  unsigned short* h2wT   = (unsigned short*)carve((size_t)1024 * 1088 * 2);          // (1088,1024)
  float*          xh     = (float*)         carve((size_t)B_ * 512 * 4);
  unsigned short* xhn    = (unsigned short*)carve((size_t)B_ * 512 * 2);
  unsigned short* h1buf  = (unsigned short*)carve((size_t)B_ * 1024 * 2);
  float*          h2f    = (float*)         carve((size_t)B_ * 1088 * 4);
  (void)ws_size; (void)in_sizes; (void)n_in; (void)out_size;

  // ---- geometric front-end ----
  {
    int tot = B_ * L_ * NTOT;
    k_build_pts<<<(tot + 255) / 256, 256, 0, stream>>>(points, pts);
  }
  k_fps<<<B_ * NF, 256, 0, stream>>>(pts, a_idx, a_xyz);
  k_group<<<B_ * NF * MM, 256, 0, stream>>>(pts, a_xyz, conv_d_w, conv_f_w, pos_w, pos_b, xbuf);

  // ---- transpose+convert weights to bf16 (K x N -> N x K) ----
  auto cvtT = [&](const float* s, unsigned short* d, int K, int N, int nz) {
    k_cvt_t<<<dim3((N + 31) / 32, (K + 31) / 32, nz), 256, 0, stream>>>(s, d, K, N);
  };
  cvtT(qkv_w,      wqT,  512, 1536, DEPTH_);
  cvtT(attn_out_w, woT,  512, 512,  DEPTH_);
  cvtT(ff1_w,      w1T,  512, 1024, DEPTH_);
  cvtT(ff2_w,      w2T,  1024, 512, DEPTH_);
  cvtT(head1_w,    h1wT, 512, 1024, 1);
  cvtT(head2_w,    h2wT, 1024, 1088, 1);

  // ---- transformer ----
  for (int l = 0; l < DEPTH_; ++l) {
    k_ln<<<(int)ROWS, 256, 0, stream>>>(xbuf, ln1_g + l * DIM_, ln1_b + l * DIM_, xn);

    // QKV: (6240x512) @ (512x1536) -> bf16 qkv   (W^T: 1536x512)
    k_gemm<<<dim3(12, 49, 1), 256, 0, stream>>>(
        xn, 0, 0, 512,
        wqT + (long long)l * 512 * 1536, 0, 0, 512, 1,
        nullptr, nullptr, 0, 0,
        nullptr, 0, 0,
        qkvb, 0, 0, 1536,
        (int)ROWS, 1536, 512, 1, 1.0f, 0);

    // scores[z=(b,h)] = 0.125 * Q @ K^T : (780x64)@(64x780) -> f32
    k_gemm<<<dim3(7, 7, BH), 256, 0, stream>>>(
        qkvb, (long long)SEQ * 1536, 64, 1536,
        qkvb + 512, (long long)SEQ * 1536, 64, 1536, 1 /*W^T*/,
        nullptr, nullptr, 0, 0,
        scores, 8 * SS, SS,
        nullptr, 0, 0, SEQ,
        SEQ, SEQ, 64, HEADS_, 0.125f, 0);

    k_softmax<<<BH * SEQ, 256, 0, stream>>>(scores, probs);

    // o[z] = P @ V : (780x780)@(780x64) -> bf16 into (B,SEQ,512) head slice
    // (V is row-major inside qkvb -> the one remaining gather-path GEMM)
    k_gemm<<<dim3(1, 7, BH), 256, 0, stream>>>(
        probs, 8 * SS, SS, SEQ,
        qkvb + 1024, (long long)SEQ * 1536, 64, 1536, 0,
        nullptr, nullptr, 0, 0,
        nullptr, 0, 0,
        obuf, (long long)SEQ * 512, 64, 512,
        SEQ, 64, SEQ, HEADS_, 1.0f, 0);

    // x = x + o @ Wo + bo   (Wo^T: 512x512)
    k_gemm<<<dim3(4, 49, 1), 256, 0, stream>>>(
        obuf, 0, 0, 512,
        woT + (long long)l * 512 * 512, 0, 0, 512, 1,
        attn_out_b + l * 512,
        xbuf, 0, 0,
        xbuf, 0, 0,
        nullptr, 0, 0, 512,
        (int)ROWS, 512, 512, 1, 1.0f, 0);

    k_ln<<<(int)ROWS, 256, 0, stream>>>(xbuf, ln2_g + l * DIM_, ln2_b + l * DIM_, xn);

    // ffh = gelu(xn @ W1 + b1) -> bf16   (W1^T: 1024x512)
    k_gemm<<<dim3(8, 49, 1), 256, 0, stream>>>(
        xn, 0, 0, 512,
        w1T + (long long)l * 512 * 1024, 0, 0, 512, 1,
        ff1_b + l * 1024,
        nullptr, 0, 0,
        nullptr, 0, 0,
        ffh, 0, 0, 1024,
        (int)ROWS, 1024, 512, 1, 1.0f, 1);

    // x = x + ffh @ W2 + b2   (W2^T: 512x1024)
    k_gemm<<<dim3(4, 49, 1), 256, 0, stream>>>(
        ffh, 0, 0, 1024,
        w2T + (long long)l * 1024 * 512, 0, 0, 1024, 1,
        ff2_b + l * 512,
        xbuf, 0, 0,
        xbuf, 0, 0,
        nullptr, 0, 0, 512,
        (int)ROWS, 512, 1024, 1, 1.0f, 0);
  }

  // ---- head ----
  k_seqmax<<<(B_ * DIM_ + 255) / 256, 256, 0, stream>>>(xbuf, xh);
  k_ln<<<B_, 256, 0, stream>>>(xh, head_ln_g, head_ln_b, xhn);

  k_gemm<<<dim3(8, 1, 1), 256, 0, stream>>>(
      xhn, 0, 0, 512,
      h1wT, 0, 0, 512, 1,
      head1_b,
      nullptr, 0, 0,
      nullptr, 0, 0,
      h1buf, 0, 0, 1024,
      B_, 1024, 512, 1, 1.0f, 1);

  k_gemm<<<dim3(9, 1, 1), 256, 0, stream>>>(
      h1buf, 0, 0, 1024,
      h2wT, 0, 0, 1024, 1,
      head2_b,
      nullptr, 0, 0,
      h2f, 0, 0,
      nullptr, 0, 0, 1088,
      B_, 1088, 1024, 1, 1.0f, 0);

  k_head_out<<<B_ * 17, 64, 0, stream>>>(h2f, (float*)d_out);
}
